// GaussianQueryModule_36567351558251
// MI455X (gfx1250) — matrix operand order repaired and checked by reading, not compiled, not run on hardware
//
#include <hip/hip_runtime.h>

typedef __attribute__((ext_vector_type(16))) _Float16 v16h;
typedef __attribute__((ext_vector_type(8)))  float    v8f;
typedef __attribute__((ext_vector_type(4)))  unsigned int v4u;
typedef __attribute__((ext_vector_type(8)))  int      v8i;
typedef __attribute__((ext_vector_type(4)))  int      v4i;

// ---------------- WMMA helpers (CDNA5 16x16x32 f16 -> f32) ----------------

__device__ __forceinline__ v8f wmma32(v16h a, v16h b, v8f c) {
  // D = A(16x32) * B(32x16) + C
  return __builtin_amdgcn_wmma_f32_16x16x32_f16(false, a, false, b, (short)0, c,
                                                false, false);
}

// A fragment: row-major A[m][k] (lda in halves). Lane layout (ISA 7.12.2):
// m = lane&15 ; lane-half hi adds +8 to k base; element pairs contiguous in k.
__device__ __forceinline__ v16h loadA_rm(const _Float16* A, int lda, int m0,
                                         int k0, int lane) {
  int m = m0 + (lane & 15);
  const _Float16* row = A + (size_t)m * lda + k0 + ((lane >> 4) & 1) * 8;
  v16h a;
#pragma unroll
  for (int v = 0; v < 8; ++v) {
    int kb = (v < 4) ? (2 * v) : (16 + 2 * (v - 4));
    a[2 * v]     = row[kb];
    a[2 * v + 1] = row[kb + 1];
  }
  return a;
}

__device__ __forceinline__ v16h loadA_rm_guard(const _Float16* A, int lda,
                                               int mMax, int m0, int k0,
                                               int lane) {
  int m = m0 + (lane & 15);
  v16h a;
  if (m >= mMax) {
#pragma unroll
    for (int e = 0; e < 16; ++e) a[e] = (_Float16)0.0f;
    return a;
  }
  const _Float16* row = A + (size_t)m * lda + k0 + ((lane >> 4) & 1) * 8;
#pragma unroll
  for (int v = 0; v < 8; ++v) {
    int kb = (v < 4) ? (2 * v) : (16 + 2 * (v - 4));
    a[2 * v]     = row[kb];
    a[2 * v + 1] = row[kb + 1];
  }
  return a;
}

// A fragment for implicit-GEMM conv3x3 from an LDS halo tile:
// pixel m -> (m/tileW, m%tileW); k chunk = tap t (dy*3+dx), channels ci0..+31.
__device__ __forceinline__ v16h loadA_halo(const _Float16* halo, int haloW,
                                           int tileW, int m0, int t, int ci0,
                                           int lane) {
  int m  = m0 + (lane & 15);
  int py = m / tileW + t / 3;
  int px = m % tileW + t % 3;
  const _Float16* row = halo + ((py * haloW + px) << 6) + ci0 + ((lane >> 4) & 1) * 8;
  v16h a;
#pragma unroll
  for (int v = 0; v < 8; ++v) {
    int kb = (v < 4) ? (2 * v) : (16 + 2 * (v - 4));
    a[2 * v]     = row[kb];
    a[2 * v + 1] = row[kb + 1];
  }
  return a;
}

// B fragment from N-major weights Bt[n][k] (ldk halves): lane n = n0+(lane&15),
// k = k0 + 8*(lane>=16) + (e<8 ? e : e+8)  -> matches A fragment k order.
__device__ __forceinline__ v16h loadB_nk(const _Float16* Bt, int ldk, int n0,
                                         int k0, int lane) {
  int n = n0 + (lane & 15);
  const _Float16* row = Bt + (size_t)n * ldk + k0 + ((lane >> 4) & 1) * 8;
  v16h b;
#pragma unroll
  for (int e = 0; e < 16; ++e) b[e] = row[(e < 8) ? e : (e + 8)];
  return b;
}

// ---------------- tiny utility kernels ----------------

__global__ void zero_f32(float* p, int n) {
  int i = blockIdx.x * blockDim.x + threadIdx.x;
  if (i < n) p[i] = 0.0f;
}

// conv weights [O][64][3][3] f32 -> f16 [O][576], k = (dy*3+dx)*64 + ci
__global__ void pack_conv(const float* __restrict__ src, _Float16* __restrict__ dst,
                          int O) {
  int n = O * 576;
  for (int i = blockIdx.x * blockDim.x + threadIdx.x; i < n;
       i += gridDim.x * blockDim.x) {
    int o = i / 576, k = i % 576;
    int t = k >> 6, ci = k & 63;
    dst[i] = (_Float16)src[(o * 64 + ci) * 9 + t];
  }
}

// dense [Nv][K] f32 -> f16 [N][K] with zero padding rows Nv..N-1
__global__ void pack_mat(const float* __restrict__ src, _Float16* __restrict__ dst,
                         int N, int K, int Nv) {
  int n = N * K;
  for (int i = blockIdx.x * blockDim.x + threadIdx.x; i < n;
       i += gridDim.x * blockDim.x) {
    int r = i / K, c = i % K;
    dst[i] = (_Float16)((r < Nv) ? src[r * K + c] : 0.0f);
  }
}

// feat [2,64,112,112] f32 NCHW -> f16 NHWC
__global__ void to_nhwc(const float* __restrict__ src, _Float16* __restrict__ dst) {
  int n = 2 * 112 * 112 * 64;
  for (int i = blockIdx.x * blockDim.x + threadIdx.x; i < n;
       i += gridDim.x * blockDim.x) {
    int c = i & 63;
    int r = i >> 6;
    int x = r % 112; r /= 112;
    int y = r % 112;
    int b = r / 112;
    dst[i] = (_Float16)src[((b * 64 + c) * 112 + y) * 112 + x];
  }
}

// ---------------- classifier conv3x3 64->64 + bias + relu (implicit GEMM) ----------------
__global__ __launch_bounds__(256) void conv3x3_relu64(
    const _Float16* __restrict__ in, const _Float16* __restrict__ Wt,
    const float* __restrict__ bias, _Float16* __restrict__ out) {
  __shared__ _Float16 halo[10 * 10 * 64];
  int bx = blockIdx.x, by = blockIdx.y, b = blockIdx.z;
  int tid = threadIdx.x, lane = tid & 31, w = tid >> 5;

  for (int i = tid; i < 10 * 10 * 64; i += 256) {
    int hy = i / 640, hx = (i % 640) >> 6, c = i & 63;
    int Y = by * 8 + hy - 1, X = bx * 8 + hx - 1;
    _Float16 v = (_Float16)0.0f;
    if (Y >= 0 && Y < 112 && X >= 0 && X < 112)
      v = in[((size_t)((b * 112 + Y) * 112 + X) << 6) + c];
    halo[i] = v;
  }
  __syncthreads();

  int mBase = (w & 3) * 16;      // 4 M-tiles cover 64 pixels
  int nBase = (w >> 2) * 32;     // 2 wave-cols x 32 = 64 out channels
  v8f a0 = {}, a1 = {};
  for (int t = 0; t < 9; ++t) {
    for (int ci0 = 0; ci0 < 64; ci0 += 32) {
      v16h af = loadA_halo(halo, 10, 8, mBase, t, ci0, lane);
      v16h b0 = loadB_nk(Wt, 576, nBase, t * 64 + ci0, lane);
      v16h b1 = loadB_nk(Wt, 576, nBase + 16, t * 64 + ci0, lane);
      a0 = wmma32(af, b0, a0);
      a1 = wmma32(af, b1, a1);
    }
  }
  int n0 = nBase + (lane & 15);
#pragma unroll
  for (int j = 0; j < 8; ++j) {
    int m = mBase + j + ((lane >> 4) << 3);
    int Y = by * 8 + (m >> 3), X = bx * 8 + (m & 7);
    size_t base = ((size_t)((b * 112 + Y) * 112 + X) << 6);
    out[base + n0]      = (_Float16)fmaxf(a0[j] + bias[n0], 0.0f);
    out[base + n0 + 16] = (_Float16)fmaxf(a1[j] + bias[n0 + 16], 0.0f);
  }
}

// ---------------- conv1x1 64->100 logits + softmax + weighted accumulation ----------------
// Activation tile staged into LDS by the Tensor Data Mover (one DMA per block):
// 2D tile, 8 NHWC rows x 1024B contiguous, row stride 14336B, 4-byte units.
__global__ __launch_bounds__(256) void cls_logits_acc(
    const _Float16* __restrict__ h1, const _Float16* __restrict__ W2,
    const float* __restrict__ b2, const float* __restrict__ sx,
    const float* __restrict__ sy, const float* __restrict__ op,
    const float* __restrict__ rh, float* __restrict__ acc) {
  __shared__ _Float16 At[64 * 64];
  __shared__ float lg[64 * 112];
  int bx = blockIdx.x, by = blockIdx.y, b = blockIdx.z;
  int tid = threadIdx.x, lane = tid & 31, w = tid >> 5;

  if (w == 0) {
    // D# group0: count=1 | lds_addr | global_addr(57b) | type=2
    unsigned lds = (unsigned)(size_t)(void*)At;  // LDS address: low 32 bits of the pointer
    unsigned long long ga = (unsigned long long)(size_t)(const void*)(
        h1 + ((size_t)((b * 112 + by * 8) * 112 + bx * 8) << 6));
    v4u g0;
    g0[0] = 1u;
    g0[1] = lds;
    g0[2] = (unsigned)ga;
    g0[3] = (unsigned)(ga >> 32) | 0x80000000u;  // type=2 ("image")
    // D# group1: data_size=2 (4B), tensor dims 2^30 (interior tile, no OOB),
    // tile_dim0=256 units (1024B), tile_dim1=8 rows, dim0 stride=3584 units.
    v8i g1;
    g1[0] = 0x00020000;                       // data_size=2
    g1[1] = 0;                                // tensor_dim0 lo16 << 16
    g1[2] = 0x4000;                           // tensor_dim0 hi16 | dim1 lo16<<16
    g1[3] = (int)(0x4000u | (256u << 16));    // dim1 hi16 | tile_dim0=256
    g1[4] = 8;                                // tile_dim1=8, tile_dim2=0
    g1[5] = 3584;                             // tensor_dim0_stride (units)
    g1[6] = 0;
    g1[7] = 0;
    v4i gz4 = {0, 0, 0, 0};
    v8i gz8 = {0, 0, 0, 0, 0, 0, 0, 0};
    __builtin_amdgcn_tensor_load_to_lds(g0, g1, gz4, gz4, gz8, 0);
    __builtin_amdgcn_s_wait_tensorcnt(0);
  }
  __syncthreads();

  if (w < 7) {
    int nBase = w * 16;
    for (int mt = 0; mt < 4; ++mt) {
      v8f c = {};
      for (int k0 = 0; k0 < 64; k0 += 32) {
        v16h af = loadA_rm(At, 64, mt * 16, k0, lane);
        v16h bf = loadB_nk(W2, 64, nBase, k0, lane);
        c = wmma32(af, bf, c);
      }
#pragma unroll
      for (int j = 0; j < 8; ++j) {
        int m = mt * 16 + j + ((lane >> 4) << 3);
        lg[m * 112 + nBase + (lane & 15)] = c[j];
      }
    }
  }
  __syncthreads();

  if (tid < 64) {
    int m = tid;
    int Y = by * 8 + (m >> 3), X = bx * 8 + (m & 7);
    int p = (Y % 7) * 7 + (X % 7);
    float mx = -1e30f;
    for (int g = 0; g < 100; ++g) mx = fmaxf(mx, lg[m * 112 + g] + b2[g]);
    float se = 0.f, d0 = 0.f, d1 = 0.f, d2 = 0.f, d3 = 0.f;
    for (int g = 0; g < 100; ++g) {
      float e = __expf(lg[m * 112 + g] + b2[g] - mx);
      se += e;
      d0 += e * sx[g]; d1 += e * sy[g]; d2 += e * op[g]; d3 += e * rh[g];
    }
    float inv = 1.0f / se;
    atomicAdd(&acc[p * 4 + 0], d0 * inv);
    atomicAdd(&acc[p * 4 + 1], d1 * inv);
    atomicAdd(&acc[p * 4 + 2], d2 * inv);
    atomicAdd(&acc[p * 4 + 3], d3 * inv);
  }
}

// ---------------- build translated gaussian masks (once, tiny) ----------------
__global__ void build_trans(const float* __restrict__ acc,
                            _Float16* __restrict__ transA,
                            float* __restrict__ opF) {
  int p = threadIdx.x;
  if (p >= 49) {
    if (p < 64)
      for (int m = 0; m < 196; ++m) transA[m * 64 + p] = (_Float16)0.0f;
    return;
  }
  const float BLinv = 1.0f / 512.0f;
  float s_x = acc[p * 4 + 0] * BLinv;
  float s_y = acc[p * 4 + 1] * BLinv;
  float o   = acc[p * 4 + 2] * BLinv;
  float r   = acc[p * 4 + 3] * BLinv;
  float a  = s_x * s_x + 1e-5f, d = s_y * s_y + 1e-5f;
  float bc = r * s_x * s_y;
  float det = a * d - bc * bc;
  float i00 = d / det, i01 = -bc / det, i11 = a / det;

  float kp[14][14];
  for (int y = 0; y < 14; ++y)
    for (int x = 0; x < 14; ++x) kp[y][x] = 0.0f;
  for (int h = 0; h < 5; ++h)
    for (int wq = 0; wq < 5; ++wq) {
      float yv = -5.0f + 2.5f * h, xv = -5.0f + 2.5f * wq;
      float z = -0.5f * (i00 * yv * yv + 2.0f * i01 * yv * xv + i11 * xv * xv);
      kp[h + 4][wq + 4] = __expf(z);  // kern / kern.max == exp(z)
    }
  int py = p / 7, px = p % 7;
  float cn0 = (0.5f - (float)py / 7.0f) * 2.0f;
  float cn1 = (0.5f - (float)px / 7.0f) * 2.0f;
  float dxs = cn0 * 13.0f * 0.5f;  // column shift (from cn[:,0])
  float dys = cn1 * 13.0f * 0.5f;  // row shift    (from cn[:,1])
  for (int h = 0; h < 14; ++h)
    for (int wq = 0; wq < 14; ++wq) {
      float rr = (float)h + dys, cc = (float)wq + dxs;
      int r0 = (int)floorf(rr), c0 = (int)floorf(cc);
      float fr = rr - r0, fc = cc - c0;
      float v = 0.0f;
#pragma unroll
      for (int dr = 0; dr < 2; ++dr)
#pragma unroll
        for (int dc = 0; dc < 2; ++dc) {
          int R = r0 + dr, C = c0 + dc;
          float g = (R >= 0 && R < 14 && C >= 0 && C < 14) ? kp[R][C] : 0.0f;
          v += g * (dr ? fr : 1.0f - fr) * (dc ? fc : 1.0f - fc);
        }
      transA[(h * 14 + wq) * 64 + p] = (_Float16)v;
    }
  opF[p] = o;
}

// ---------------- splat + fold: gf[b,Y,X,c] (NHWC f16) ----------------
__global__ __launch_bounds__(256) void splat_fold(
    const _Float16* __restrict__ featH, const float* __restrict__ opF,
    const _Float16* __restrict__ transA, _Float16* __restrict__ gf) {
  __shared__ _Float16 Bt[64 * 64];  // [c][kpad]
  int tc = blockIdx.x, tr = blockIdx.y, b = blockIdx.z;
  int tid = threadIdx.x, lane = tid & 31, w = tid >> 5;

  for (int i = tid; i < 64 * 64; i += 256) {
    int c = i >> 6, p = i & 63;
    _Float16 v = (_Float16)0.0f;
    if (p < 49) {
      int Y = tr * 7 + p / 7, X = tc * 7 + p % 7;
      v = (_Float16)((float)featH[((size_t)((b * 112 + Y) * 112 + X) << 6) + c] * opF[p]);
    }
    Bt[c * 64 + p] = v;  // N-major [n=c][k=p]
  }
  __syncthreads();

  int nBase = (w & 3) * 16;
  int wr = w >> 2;
  for (int mt = wr; mt < 13; mt += 2) {
    v8f c = {};
    for (int k0 = 0; k0 < 64; k0 += 32) {
      v16h af = loadA_rm_guard(transA, 64, 196, mt * 16, k0, lane);
      v16h bf = loadB_nk(Bt, 64, nBase, k0, lane);
      c = wmma32(af, bf, c);
    }
#pragma unroll
    for (int j = 0; j < 8; ++j) {
      int m = mt * 16 + j + ((lane >> 4) << 3);
      if (m < 196) {
        int hh = m / 14, ww = m % 14;
        int Y = tr * 14 + hh, X = tc * 14 + ww;
        float v = fminf(fmaxf(c[j], 0.0f), 1.0f);
        gf[((size_t)((b * 224 + Y) * 224 + X) << 6) + nBase + (lane & 15)] = (_Float16)v;
      }
    }
  }
}

// ---------------- fused LTE tail: coef/freq conv3x3(64->256) + phase + 3-layer MLP ----------------
__global__ __launch_bounds__(256) void lte_query(
    const _Float16* __restrict__ gf,
    const _Float16* __restrict__ Wc, const float* __restrict__ cb,
    const _Float16* __restrict__ Wf, const float* __restrict__ fb,
    const float* __restrict__ coord, const float* __restrict__ cell,
    const float* __restrict__ phw,
    const _Float16* __restrict__ W0, const float* __restrict__ b0,
    const _Float16* __restrict__ W1, const float* __restrict__ b1,
    const _Float16* __restrict__ W2, const float* __restrict__ b2,
    float* __restrict__ out) {
  __shared__ _Float16 sm[3840 + 3 * 8192];  // 56,832 bytes
  _Float16* halo  = sm;                 // 10x6x64
  _Float16* coefL = sm + 3840;          // [32][256]
  _Float16* freqL = sm + 3840 + 8192;   // [32][256]
  _Float16* x0    = sm + 3840 + 16384;  // [32][256]
  _Float16* x1    = coefL;              // reuse after transform
  _Float16* x2    = freqL;              // reuse after layer 1

  int bx = blockIdx.x, by = blockIdx.y, b = blockIdx.z;
  int tid = threadIdx.x, lane = tid & 31, w = tid >> 5;

  // warm L2 with the weight streams (global_prefetch_b8; fetches whole lines)
  if (tid == 0) {
    __builtin_prefetch(Wc, 0, 1);
    __builtin_prefetch(Wf, 0, 1);
    __builtin_prefetch(W0, 0, 1);
    __builtin_prefetch(W1, 0, 1);
  }

  // halo load (conv pad=1, zeros outside)
  for (int i = tid; i < 10 * 6 * 64; i += 256) {
    int hy = i / 384, hx = (i % 384) >> 6, c = i & 63;
    int Y = by * 8 + hy - 1, X = bx * 4 + hx - 1;
    _Float16 v = (_Float16)0.0f;
    if (Y >= 0 && Y < 224 && X >= 0 && X < 224)
      v = gf[((size_t)((b * 224 + Y) * 224 + X) << 6) + c];
    halo[i] = v;
  }
  __syncthreads();

  // two convs: 32 N-chunks of 16 (0..15 coef, 16..31 freq); 2 M-tiles x 4 wave-cols
  int mBase = (w & 1) * 16;
  int wc = w >> 1;
  for (int nc = wc; nc < 32; nc += 4) {
    const _Float16* Wt  = (nc < 16) ? Wc : Wf;
    const float* bias   = (nc < 16) ? cb : fb;
    _Float16* dst       = (nc < 16) ? coefL : freqL;
    int n0 = (nc & 15) * 16;
    v8f c = {};
    for (int t = 0; t < 9; ++t)
      for (int ci0 = 0; ci0 < 64; ci0 += 32) {
        v16h af = loadA_halo(halo, 6, 4, mBase, t, ci0, lane);
        v16h bf = loadB_nk(Wt, 576, n0, t * 64 + ci0, lane);
        c = wmma32(af, bf, c);
      }
#pragma unroll
    for (int j = 0; j < 8; ++j) {
      int m = mBase + j + ((lane >> 4) << 3);
      int n = n0 + (lane & 15);
      dst[m * 256 + n] = (_Float16)(c[j] + bias[n]);
    }
  }
  __syncthreads();

  // phase transform: x0[c] = q_coef[c] * {cos,sin}(pi*(freq·rel + rc@phw^T))
  {
    int pix = tid >> 3, part = tid & 7;
    int Y = by * 8 + (pix >> 2), X = bx * 4 + (pix & 3);
    int q = Y * 224 + X;
    size_t qb = (size_t)(b * 50176 + q) * 2;
    float c0 = coord[qb + 0], c1 = coord[qb + 1];
    float e0 = cell[qb + 0],  e1 = cell[qb + 1];
    float yif = fminf(fmaxf(rintf((c0 + 1.0f) * 112.0f - 0.5f), 0.0f), 223.0f);
    float xif = fminf(fmaxf(rintf((c1 + 1.0f) * 112.0f - 0.5f), 0.0f), 223.0f);
    float ycen = -1.0f + 1.0f / 224.0f + (2.0f / 224.0f) * yif;
    float xcen = -1.0f + 1.0f / 224.0f + (2.0f / 224.0f) * xif;
    float rel0 = (c0 - ycen) * 224.0f, rel1 = (c1 - xcen) * 224.0f;
    float rc0 = e0 * 224.0f, rc1 = e1 * 224.0f;
    for (int j = part * 16; j < part * 16 + 16; ++j) {
      float f0 = (float)freqL[pix * 256 + 2 * j];
      float f1 = (float)freqL[pix * 256 + 2 * j + 1];
      float ph = rc0 * phw[2 * j] + rc1 * phw[2 * j + 1];
      float qs = 3.14159265358979f * (f0 * rel0 + f1 * rel1 + ph);
      float s, cth;
      __sincosf(qs, &s, &cth);
      x0[pix * 256 + j]       = (_Float16)((float)coefL[pix * 256 + j] * cth);
      x0[pix * 256 + 128 + j] = (_Float16)((float)coefL[pix * 256 + 128 + j] * s);
    }
  }
  __syncthreads();

  // MLP layer 1: x1 = relu(x0 @ W0^T + b0)
  for (int nt = wc; nt < 16; nt += 4) {
    int n0 = nt * 16;
    v8f c = {};
    for (int k0 = 0; k0 < 256; k0 += 32) {
      v16h af = loadA_rm(x0, 256, mBase, k0, lane);
      v16h bf = loadB_nk(W0, 256, n0, k0, lane);
      c = wmma32(af, bf, c);
    }
#pragma unroll
    for (int j = 0; j < 8; ++j) {
      int m = mBase + j + ((lane >> 4) << 3);
      int n = n0 + (lane & 15);
      x1[m * 256 + n] = (_Float16)fmaxf(c[j] + b0[n], 0.0f);
    }
  }
  __syncthreads();

  // MLP layer 2: x2 = relu(x1 @ W1^T + b1)
  for (int nt = wc; nt < 16; nt += 4) {
    int n0 = nt * 16;
    v8f c = {};
    for (int k0 = 0; k0 < 256; k0 += 32) {
      v16h af = loadA_rm(x1, 256, mBase, k0, lane);
      v16h bf = loadB_nk(W1, 256, n0, k0, lane);
      c = wmma32(af, bf, c);
    }
#pragma unroll
    for (int j = 0; j < 8; ++j) {
      int m = mBase + j + ((lane >> 4) << 3);
      int n = n0 + (lane & 15);
      x2[m * 256 + n] = (_Float16)fmaxf(c[j] + b1[n], 0.0f);
    }
  }
  __syncthreads();

  // final layer: out = x2 @ W2^T + b2 (N padded 3->16)
  if (wc == 0) {
    v8f c = {};
    for (int k0 = 0; k0 < 256; k0 += 32) {
      v16h af = loadA_rm(x2, 256, mBase, k0, lane);
      v16h bf = loadB_nk(W2, 256, 0, k0, lane);
      c = wmma32(af, bf, c);
    }
    int n = lane & 15;
    if (n < 3) {
#pragma unroll
      for (int j = 0; j < 8; ++j) {
        int m = mBase + j + ((lane >> 4) << 3);
        int Y = by * 8 + (m >> 2), X = bx * 4 + (m & 3);
        int q = Y * 224 + X;
        out[(size_t)(b * 50176 + q) * 3 + n] = c[j] + b2[n];
      }
    }
  }
}

// ---------------- host launch ----------------

extern "C" void kernel_launch(void* const* d_in, const int* in_sizes, int n_in,
                              void* d_out, int out_size, void* d_ws, size_t ws_size,
                              hipStream_t stream) {
  (void)in_sizes; (void)n_in; (void)out_size; (void)ws_size;
  const float* feat    = (const float*)d_in[0];
  const float* coord   = (const float*)d_in[1];
  const float* cell    = (const float*)d_in[2];
  const float* cls_w1  = (const float*)d_in[3];
  const float* cls_b1  = (const float*)d_in[4];
  const float* cls_w2  = (const float*)d_in[5];
  const float* cls_b2  = (const float*)d_in[6];
  const float* sigx    = (const float*)d_in[7];
  const float* sigy    = (const float*)d_in[8];
  const float* opac    = (const float*)d_in[9];
  const float* rho     = (const float*)d_in[10];
  const float* coef_w  = (const float*)d_in[11];
  const float* coef_b  = (const float*)d_in[12];
  const float* freq_w  = (const float*)d_in[13];
  const float* freq_b  = (const float*)d_in[14];
  const float* phase_w = (const float*)d_in[15];
  const float* mw0 = (const float*)d_in[16];
  const float* mb0 = (const float*)d_in[17];
  const float* mw1 = (const float*)d_in[18];
  const float* mb1 = (const float*)d_in[19];
  const float* mw2 = (const float*)d_in[20];
  const float* mb2 = (const float*)d_in[21];
  float* out = (float*)d_out;

  char* ws = (char*)d_ws;
  size_t off = 0;
  auto take = [&](size_t bytes) -> char* {
    char* p = ws + off;
    off = (off + bytes + 255) & ~(size_t)255;
    return p;
  };
  _Float16* wB1    = (_Float16*)take((size_t)64 * 576 * 2);
  _Float16* wB2    = (_Float16*)take((size_t)112 * 64 * 2);
  _Float16* wCo    = (_Float16*)take((size_t)256 * 576 * 2);
  _Float16* wFr    = (_Float16*)take((size_t)256 * 576 * 2);
  _Float16* wM0    = (_Float16*)take((size_t)256 * 256 * 2);
  _Float16* wM1    = (_Float16*)take((size_t)256 * 256 * 2);
  _Float16* wM2    = (_Float16*)take((size_t)16 * 256 * 2);
  _Float16* featH  = (_Float16*)take((size_t)2 * 112 * 112 * 64 * 2);
  _Float16* h1     = (_Float16*)take((size_t)2 * 112 * 112 * 64 * 2);
  _Float16* gf     = (_Float16*)take((size_t)2 * 224 * 224 * 64 * 2);
  float*    accP   = (float*)take((size_t)49 * 4 * 4);
  float*    opF    = (float*)take((size_t)49 * 4);
  _Float16* transA = (_Float16*)take((size_t)196 * 64 * 2);

  zero_f32<<<1, 256, 0, stream>>>(accP, 196);
  pack_conv<<<144, 256, 0, stream>>>(cls_w1, wB1, 64);
  pack_conv<<<576, 256, 0, stream>>>(coef_w, wCo, 256);
  pack_conv<<<576, 256, 0, stream>>>(freq_w, wFr, 256);
  pack_mat<<<28, 256, 0, stream>>>(cls_w2, wB2, 112, 64, 100);
  pack_mat<<<256, 256, 0, stream>>>(mw0, wM0, 256, 256, 256);
  pack_mat<<<256, 256, 0, stream>>>(mw1, wM1, 256, 256, 256);
  pack_mat<<<16, 256, 0, stream>>>(mw2, wM2, 16, 256, 3);
  to_nhwc<<<6272, 256, 0, stream>>>(feat, featH);

  dim3 g1(14, 14, 2);
  conv3x3_relu64<<<g1, 256, 0, stream>>>(featH, wB1, cls_b1, h1);
  cls_logits_acc<<<g1, 256, 0, stream>>>(h1, wB2, cls_b2, sigx, sigy, opac, rho, accP);
  build_trans<<<1, 64, 0, stream>>>(accP, transA, opF);

  dim3 g2(16, 16, 2);
  splat_fold<<<g2, 256, 0, stream>>>(featH, opF, transA, gf);

  dim3 g3(56, 28, 2);
  lte_query<<<g3, 256, 0, stream>>>(gf, wCo, coef_b, wFr, freq_b, coord, cell,
                                    phase_w, wM0, mb0, wM1, mb1, wM2, mb2, out);
}
